// SimpleGauss_54150947668144
// MI455X (gfx1250) — compile-verified, matrix operand order checked by reading, not run on hardware
//
#include <hip/hip_runtime.h>
#include <hip/hip_bf16.h>

typedef __attribute__((ext_vector_type(16))) _Float16           v16h;
typedef __attribute__((ext_vector_type(8)))  _Float16           v8h;
typedef __attribute__((ext_vector_type(8)))  float              v8f;
typedef __attribute__((ext_vector_type(4)))  unsigned long long v4u64;

#define C_CONST 8
#define NP      2048
#define KCLS    64
#define NIN     8192
#define DK      512     // C_CONST * KCLS
#define EPS_T   1e-13f

#define JROWS   64      // prototype rows per block (4 WMMA sub-tiles)
#define NJB     (NP / JROWS)     // 32 j-blocks
#define NQ      16               // i chunks of 512
#define QCOLS   (NIN / NQ)       // 512

// workspace byte offsets
#define WS_AHALF 0u         // 2048*512 halves  = 2 MB
#define WS_SUMSQ 2097152u   // 16384 floats     = 64 KB
#define WS_AROW  2162688u   // 2048 floats      = 8 KB
#define WS_PIDX  2170880u   // 8192 uint2       = 64 KB
#define WS_SPART 2236416u   // 32*16*64 floats  = 128 KB
#define WS_SFULL 2367488u   // 2048 floats      = 8 KB

// ---------------------------------------------------------------- pack idx
__global__ void k_pack_idx(const int* __restrict__ c_in, uint2* __restrict__ pidx) {
  int t = blockIdx.x * blockDim.x + threadIdx.x;
  if (t < NIN) {
    unsigned lo = 0u, hi = 0u;
#pragma unroll
    for (int c = 0; c < 4; ++c) lo |= ((unsigned)c_in[t * 8 + c] & 63u) << (8 * c);
#pragma unroll
    for (int c = 0; c < 4; ++c) hi |= ((unsigned)c_in[t * 8 + 4 + c] & 63u) << (8 * c);
    pidx[t] = make_uint2(lo, hi);
  }
}

// ---------------------------------------------------------------- softmax
__global__ void k_softmax(const float* __restrict__ c_p, _Float16* __restrict__ Ah,
                          float* __restrict__ sumsq) {
  __shared__ float red[64];
  const int b = blockIdx.x;             // b = c*NP + j
  const int c = b >> 11;
  const int j = b & (NP - 1);
  const int k = threadIdx.x;            // 0..63
  float x = c_p[(size_t)b * KCLS + k];
  red[k] = x; __syncthreads();
  for (int off = 32; off > 0; off >>= 1) {
    if (k < off) red[k] = fmaxf(red[k], red[k + off]);
    __syncthreads();
  }
  float m = red[0]; __syncthreads();
  float e = __expf(x - m);
  red[k] = e; __syncthreads();
  for (int off = 32; off > 0; off >>= 1) {
    if (k < off) red[k] += red[k + off];
    __syncthreads();
  }
  float s = red[0]; __syncthreads();
  float p = e / s;
  Ah[(size_t)j * DK + c * KCLS + k] = (_Float16)p;
  red[k] = p * p; __syncthreads();
  for (int off = 32; off > 0; off >>= 1) {
    if (k < off) red[k] += red[k + off];
    __syncthreads();
  }
  if (k == 0) sumsq[b] = red[0];
}

// ---------------------------------------------------------------- A[j] = C + sum_c sumsq
__global__ void k_arow(const float* __restrict__ sumsq, float* __restrict__ Arow) {
  int j = blockIdx.x * blockDim.x + threadIdx.x;
  if (j < NP) {
    float a = (float)C_CONST;
#pragma unroll
    for (int c = 0; c < C_CONST; ++c) a += sumsq[c * NP + j];
    Arow[j] = a;
  }
}

// ---------------------------------------------------------------- main WMMA kernel
// grid: (32 j-blocks x 16 i-chunks); block: 256 threads = 8 waves (wave32)
__global__ void __launch_bounds__(256)
k_gauss_wmma(const _Float16* __restrict__ Ah, const float* __restrict__ Arow,
             const uint2* __restrict__ pidx, const float* __restrict__ bwp,
             float* __restrict__ out, float* __restrict__ Spart) {
  const int jb  = blockIdx.x;           // j-block (64 rows)
  const int q   = blockIdx.y;           // i chunk (512 cols)
  const int j0  = jb * JROWS;
  const int tid = threadIdx.x;
  const int wv  = tid >> 5;
  const int ln  = tid & 31;
  const int row = ln & 15;              // A row within sub-tile / B column (query)
  const int hig = ln >> 4;              // lane half-group

  __shared__ __align__(16) _Float16 As[JROWS * DK];   // 64 KB A tile
  __shared__ float arL[JROWS];
  __shared__ float part[8][JROWS];

  // cooperative, coalesced A-tile load: 64 rows x 512 halves (4096 uint4)
  {
    const uint4* src = (const uint4*)(Ah + (size_t)j0 * DK);
    uint4* dst = (uint4*)As;
    for (int t = tid; t < JROWS * DK / 8; t += 256) dst[t] = src[t];
    if (tid < JROWS) arL[tid] = Arow[j0 + tid];
  }
  __syncthreads();

  const float bw  = fminf(fmaxf(bwp[0], 0.1f), 10.0f);
  const float ibw = 1.0f / bw;

  float ssum[4][8];
#pragma unroll
  for (int s = 0; s < 4; ++s)
#pragma unroll
    for (int r = 0; r < 8; ++r) ssum[s][r] = 0.0f;

  // this wave: 4 consecutive 16-wide i-tiles inside its chunk
  const int it0 = q * (QCOLS / 16) + wv * 4;
  for (int it = it0; it < it0 + 4; ++it) {
    // Barrier = shared-memory fence: stops LICM from hoisting the LDS
    // A-fragment loads out of this loop (which caused scratch spills).
    // Uniform trip count across the block, so this is legal.
    __syncthreads();

    const int i0 = it * 16;
    const uint2 pk = pidx[i0 + row];
    int idxs[8];
#pragma unroll
    for (int c = 0; c < 4; ++c) {
      idxs[c]     = (int)((pk.x >> (8 * c)) & 63u);
      idxs[4 + c] = (int)((pk.y >> (8 * c)) & 63u);
    }

    v8f acc[4];
#pragma unroll
    for (int s = 0; s < 4; ++s) acc[s] = (v8f){};

#pragma unroll
    for (int t = 0; t < 16; ++t) {
      const int k0 = t * 32 + hig * 8;
      const int cc = t >> 1;                       // concept of this 32-wide K slab
      // one-hot positions inside this lane's two 8-half chunks
      const unsigned ur1 = (unsigned)(idxs[cc] - (k0 & 63));
      const unsigned ur2 = (unsigned)(idxs[cc] - ((k0 + 16) & 63));
      // 128-bit one-hot chunk = two u64 halves; AMD 64-bit shift uses amount&63,
      // so one shifted constant serves both halves, selected by range compares.
      const unsigned long long m1 = 0x3C00ull << ((ur1 * 16u) & 63u);
      const unsigned long long m2 = 0x3C00ull << ((ur2 * 16u) & 63u);
      union { v4u64 u; v16h h; } bu;
      bu.u.x = (ur1 < 4u)      ? m1 : 0ull;
      bu.u.y = (ur1 - 4u < 4u) ? m1 : 0ull;
      bu.u.z = (ur2 < 4u)      ? m2 : 0ull;
      bu.u.w = (ur2 - 4u < 4u) ? m2 : 0ull;
      const v16h b = bu.h;

      // reuse this B fragment across 4 prototype sub-tiles
#pragma unroll
      for (int s = 0; s < 4; ++s) {
        const _Float16* ap = &As[(s * 16 + row) * DK + k0];
        v8h lo = *(const v8h*)ap;
        v8h hi = *(const v8h*)(ap + 16);
        v16h a = __builtin_shufflevector(lo, hi, 0, 1, 2, 3, 4, 5, 6, 7,
                                         8, 9, 10, 11, 12, 13, 14, 15);
        acc[s] = __builtin_amdgcn_wmma_f32_16x16x32_f16(false, a, false, b,
                                                        (short)0, acc[s], false, false);
      }
    }

    // fused epilogue: w = exp((2G - A[j]) / bw), store + accumulate row sums
#pragma unroll
    for (int s = 0; s < 4; ++s)
#pragma unroll
      for (int r = 0; r < 8; ++r) {
        float w = __expf((2.0f * acc[s][r] - arL[s * 16 + r + 8 * hig]) * ibw);
        ssum[s][r] += w;
        out[(size_t)(j0 + s * 16 + r + 8 * hig) * NIN + (i0 + row)] = w;
      }
  }

  // deterministic xor-tree reduction across the 16 N-lanes of each half-group
#pragma unroll
  for (int s = 0; s < 4; ++s)
#pragma unroll
    for (int r = 0; r < 8; ++r) {
#pragma unroll
      for (int off = 1; off < 16; off <<= 1)
        ssum[s][r] += __shfl_xor(ssum[s][r], off, 32);
    }
  if (row == 0) {
#pragma unroll
    for (int s = 0; s < 4; ++s)
#pragma unroll
      for (int r = 0; r < 8; ++r) part[wv][s * 16 + r + 8 * hig] = ssum[s][r];
  }
  __syncthreads();
  if (tid < JROWS) {                    // fixed-order sum over the 8 waves
    float s = 0.0f;
    for (int w = 0; w < 8; ++w) s += part[w][tid];
    Spart[((size_t)jb * NQ + q) * JROWS + tid] = s;
  }
}

// ---------------------------------------------------------------- assemble S[j]
__global__ void k_sfinal(const float* __restrict__ Spart, float* __restrict__ S) {
  int j = blockIdx.x * blockDim.x + threadIdx.x;
  if (j < NP) {
    const int jb = j / JROWS, m = j % JROWS;
    float s = 0.0f;
#pragma unroll
    for (int qq = 0; qq < NQ; ++qq) s += Spart[((size_t)jb * NQ + qq) * JROWS + m];
    S[j] = s;
  }
}

// ---------------------------------------------------------------- normalize
__global__ void k_norm(float* __restrict__ out, const float* __restrict__ S) {
  size_t g = (size_t)blockIdx.x * blockDim.x + threadIdx.x;
  const int j = (int)(g >> 13);         // NIN = 2^13
  const float s = S[j];
  const float v = out[g];
  out[g] = (s < EPS_T) ? 0.0f : v / s;
}

// ---------------------------------------------------------------- launch
extern "C" void kernel_launch(void* const* d_in, const int* in_sizes, int n_in,
                              void* d_out, int out_size, void* d_ws, size_t ws_size,
                              hipStream_t stream) {
  (void)in_sizes; (void)n_in; (void)out_size; (void)ws_size;
  const float* c_p  = (const float*)d_in[0];
  const float* bwp  = (const float*)d_in[1];
  const int*   c_in = (const int*)d_in[2];
  float* out = (float*)d_out;

  char* ws = (char*)d_ws;
  _Float16* Ah    = (_Float16*)(ws + WS_AHALF);
  float*    sumsq = (float*)(ws + WS_SUMSQ);
  float*    Arow  = (float*)(ws + WS_AROW);
  uint2*    pidx  = (uint2*)(ws + WS_PIDX);
  float*    Spart = (float*)(ws + WS_SPART);
  float*    S     = (float*)(ws + WS_SFULL);

  k_pack_idx<<<(NIN + 255) / 256, 256, 0, stream>>>(c_in, pidx);
  k_softmax<<<C_CONST * NP, 64, 0, stream>>>(c_p, Ah, sumsq);
  k_arow<<<(NP + 255) / 256, 256, 0, stream>>>(sumsq, Arow);

  dim3 grid(NJB, NQ);
  k_gauss_wmma<<<grid, 256, 0, stream>>>(Ah, Arow, pidx, bwp, out, Spart);

  k_sfinal<<<(NP + 255) / 256, 256, 0, stream>>>(Spart, S);
  k_norm<<<(NP * NIN) / 256, 256, 0, stream>>>(out, S);
}
